// HiPPOAgent_57054345560440
// MI455X (gfx1250) — matrix operand order repaired and checked by reading, not compiled, not run on hardware
//
#include <hip/hip_runtime.h>
#include <hip/hip_bf16.h>
#include <math.h>

// ---------------------------------------------------------------------------
// Types / WMMA helpers (gfx1250 wave32, v_wmma_f32_16x16x32_bf16)
// ---------------------------------------------------------------------------
typedef __attribute__((ext_vector_type(16))) __bf16       v16bf;
typedef __attribute__((ext_vector_type(8)))  float        v8f;
typedef __attribute__((ext_vector_type(4)))  unsigned int u32x4;

union FragBF {
  v16bf v;
  unsigned short u[16];
  u32x4 q[2];
};

__device__ __forceinline__ unsigned short f2bf(float x) {
  union { float f; unsigned int u; } c; c.f = x;
  unsigned int r = c.u + 0x7FFFu + ((c.u >> 16) & 1u);  // round-to-nearest-even
  return (unsigned short)(r >> 16);
}

__device__ __forceinline__ v8f wmma_bf16(v16bf a, v16bf b, v8f c) {
  return __builtin_amdgcn_wmma_f32_16x16x32_bf16(false, a, false, b, (short)0, c,
                                                 false, false);
}

// A fragment (16x32 bf16, ISA 7.12.2): lanes 0-15 row=lane, k in {0..7,16..23};
// lanes 16-31 row=lane-16, k in {8..15,24..31}.  Row-major A -> 2x b128 loads.
__device__ __forceinline__ v16bf load_a_bf(const unsigned short* __restrict__ A,
                                           int lda, int mrow, int kb, int lane) {
  int r = mrow + (lane & 15);
  int half = (lane >> 4) & 1;
  const unsigned short* p = A + (size_t)r * lda + kb + half * 8;
  FragBF f;
  f.q[0] = *reinterpret_cast<const u32x4*>(p);       // k = kb+half*8 .. +7
  f.q[1] = *reinterpret_cast<const u32x4*>(p + 16);  // k = +16 .. +23
  return f.v;
}

// B fragment (32x16 bf16): lane col = lane&15, k in {kb+half*8..+7, kb+16+half*8..+7} (table order).
// B[k][n] = W[n][k] for row-major W[N][ldw]  -> 2x b128 loads.
__device__ __forceinline__ v16bf load_b_wt(const unsigned short* __restrict__ W,
                                           int ldw, int ncol, int kb, int lane) {
  int n = ncol + (lane & 15);
  int half = (lane >> 4) & 1;
  const unsigned short* p = W + (size_t)n * ldw + kb + half * 8;
  FragBF f;
  f.q[0] = *reinterpret_cast<const u32x4*>(p);
  f.q[1] = *reinterpret_cast<const u32x4*>(p + 16);
  return f.v;
}

// ---------------------------------------------------------------------------
// Setup / conversion kernels
// ---------------------------------------------------------------------------
__global__ void cvt_f32_bf16_k(const float* __restrict__ s,
                               unsigned short* __restrict__ d, long n) {
  long i = (long)blockIdx.x * blockDim.x + threadIdx.x;
  if (i < n) d[i] = f2bf(s[i]);
}

__global__ void copy_f32_k(const float* __restrict__ s, float* __restrict__ d, int n) {
  int i = blockIdx.x * blockDim.x + threadIdx.x;
  if (i < n) d[i] = s[i];
}

__global__ void bias_sum_k(const float* __restrict__ a, const float* __restrict__ b,
                           float* __restrict__ o, int n) {
  int i = blockIdx.x * blockDim.x + threadIdx.x;
  if (i < n) o[i] = a[i] + b[i];
}

// conv weight [oc][ic][r][s] (f32) -> B-fragment-order bf16:
//   wf[step*512 + lane*16 + e]  holds  B[k][n] = w[n=lane&15][k],
//   k = step*32 + (lane>>4)*16 + e, zero beyond K.
// So per k-step each lane's 16 elements are contiguous (2x b128 loads).
__global__ void pack_conv_w_k(const float* __restrict__ w,
                              unsigned short* __restrict__ wf, int K, int Kpad) {
  int i = blockIdx.x * blockDim.x + threadIdx.x;
  int total = (Kpad >> 5) * 512;
  if (i >= total) return;
  int step = i >> 9;
  int r = i & 511;
  int lane = r >> 4, e = r & 15;
  int k = step * 32 + ((lane >> 4) << 4) + e;
  int oc = lane & 15;
  wf[i] = (k < K) ? f2bf(w[oc * K + k]) : (unsigned short)0;
}

// lofeat_w [128][520] f32 -> bf16 [128][544], zero pad
__global__ void prep_lofw_k(const float* __restrict__ w,
                            unsigned short* __restrict__ wp) {
  int i = blockIdx.x * blockDim.x + threadIdx.x;
  if (i >= 128 * 544) return;
  int n = i / 544, c = i - n * 544;
  wp[i] = (c < 520) ? f2bf(w[n * 520 + c]) : (unsigned short)0;
}

// abuf[r][512+j] = one_hot(z[r])[j] (j<8), 0 for 8<=j<32 (K padding to 544)
__global__ void prep_abuf_tail_k(const int* __restrict__ z,
                                 unsigned short* __restrict__ abuf) {
  int i = blockIdx.x * blockDim.x + threadIdx.x;
  if (i >= 16384 * 32) return;
  int r = i >> 5, j = i & 31;
  unsigned short v = 0;
  if (j < 8 && z[r] == j) v = 0x3F80;  // bf16 1.0
  abuf[(size_t)r * 544 + 512 + j] = v;
}

// ---------------------------------------------------------------------------
// Implicit-GEMM conv via WMMA.  One wave = one (image, y-row) -> 16 px x 16 oc
// tile.  Template constants + full k-unroll fold the (ic,dy,dx) decomposition
// to compile-time constants: the iy bounds test is wave-uniform (y only), the
// ix test is a single compare-select per element.  B comes from the fragment-
// order packed weights (2x b128 per lane per k-step).
// ---------------------------------------------------------------------------
template <int RS, int S, int PAD, int K, int KPAD>
__global__ void conv_wmma_k(const unsigned short* __restrict__ in,   // [N,16,16,16] bf16
                            const unsigned short* __restrict__ wf,   // fragment-packed
                            const float* __restrict__ bias,          // [16]
                            unsigned short* __restrict__ out) {      // [N,16,16,16] bf16
  int wave = blockIdx.x * (blockDim.x >> 5) + (threadIdx.x >> 5);
  int lane = threadIdx.x & 31;
  int img = wave >> 4;
  int y = wave & 15;
  int x = lane & 15;
  int half = (lane >> 4) & 1;
  const unsigned short* inb = in + ((size_t)img << 12);
  v8f acc = {0.f, 0.f, 0.f, 0.f, 0.f, 0.f, 0.f, 0.f};
#pragma unroll
  for (int kb = 0; kb < KPAD; kb += 32) {
    FragBF fa;
#pragma unroll
    for (int e = 0; e < 16; ++e) {
      // compile-time k decomposition (kb, e constants after unroll)
      int k = kb + half * 16 + e;
      unsigned short v = 0;
      if (k < K) {  // half-dependent but pattern constant per unrolled step
        int ic = k / RS;
        int rem = k - ic * RS;
        int dy = rem / S, dx = rem - dy * S;
        int iy = y + dy - PAD;   // wave-uniform bounds test
        int ix = x + dx - PAD;   // lane compare-select
        if ((unsigned)iy < 16u && (unsigned)ix < 16u)
          v = inb[(ic << 8) + (iy << 4) + ix];
      }
      fa.u[e] = v;
    }
    FragBF fb;
    const unsigned short* p = wf + (kb >> 5) * 512 + lane * 16;
    fb.q[0] = *reinterpret_cast<const u32x4*>(p);
    fb.q[1] = *reinterpret_cast<const u32x4*>(p + 8);
    acc = wmma_bf16(fa.v, fb.v, acc);
  }
  int oc = lane & 15;
  float bv = bias[oc];
#pragma unroll
  for (int v = 0; v < 8; ++v) {
    int xx = v + (half << 3);
    float val = acc[v] + bv;
    val = val > 0.f ? val : 0.01f * val;  // leaky relu
    out[((size_t)img << 12) + (oc << 8) + (y << 4) + xx] = f2bf(val);
  }
}

// ---------------------------------------------------------------------------
// Generic GEMM: out = A[MxK](bf16) @ W[NxK](bf16)^T + bias, optional leaky,
// f32 and/or bf16 outputs.  One wave = 16x64 strip (A-fragment reused 4x).
// ---------------------------------------------------------------------------
__global__ void fc_gemm_k(const unsigned short* __restrict__ A, int lda,
                          const unsigned short* __restrict__ W, int ldw,
                          const float* __restrict__ bias,
                          float* __restrict__ outF,
                          unsigned short* __restrict__ outB, int ldoB,
                          int M, int N, int K, int leaky) {
  int wave = blockIdx.x * (blockDim.x >> 5) + (threadIdx.x >> 5);
  int lane = threadIdx.x & 31;
  int nstrips = N >> 6;
  int tm = wave / nstrips;
  int ts = wave - tm * nstrips;
  if (tm >= (M >> 4)) return;  // wave-uniform: EXEC stays all-ones for WMMA
  int mrow = tm << 4;
  int half = (lane >> 4) & 1;
  v8f acc0 = {0.f,0.f,0.f,0.f,0.f,0.f,0.f,0.f};
  v8f acc1 = acc0, acc2 = acc0, acc3 = acc0;
  for (int kb = 0; kb < K; kb += 32) {
    v16bf a = load_a_bf(A, lda, mrow, kb, lane);
    acc0 = wmma_bf16(a, load_b_wt(W, ldw, (ts * 4 + 0) << 4, kb, lane), acc0);
    acc1 = wmma_bf16(a, load_b_wt(W, ldw, (ts * 4 + 1) << 4, kb, lane), acc1);
    acc2 = wmma_bf16(a, load_b_wt(W, ldw, (ts * 4 + 2) << 4, kb, lane), acc2);
    acc3 = wmma_bf16(a, load_b_wt(W, ldw, (ts * 4 + 3) << 4, kb, lane), acc3);
  }
  v8f accs[4] = {acc0, acc1, acc2, acc3};
#pragma unroll
  for (int j = 0; j < 4; ++j) {
    int n = ((ts * 4 + j) << 4) + (lane & 15);
    float bv = bias ? bias[n] : 0.f;
#pragma unroll
    for (int v = 0; v < 8; ++v) {
      int m = mrow + v + (half << 3);
      float val = accs[j][v] + bv;
      if (leaky) val = val > 0.f ? val : 0.01f * val;
      if (outF) outF[(size_t)m * N + n] = val;
      if (outB) outB[(size_t)m * ldoB + n] = f2bf(val);
    }
  }
}

// ---------------------------------------------------------------------------
// Fused LSTM step: gates = mask(h_prev) @ W_hh^T (+ precomputed gx), then
// pointwise c/h update.  Mask applied during A-fragment construction; h
// ping-pongs across launches (reads h_in, writes h_out) to avoid RAW races.
// 256 waves: tile (tm in 0..7 of B=128, tn in 0..31 of HID=512), 4 gates/wave.
// ---------------------------------------------------------------------------
__global__ void lstm_step_k(const float* __restrict__ h_in,   // [128,512]
                            float* __restrict__ h_out,        // [128,512]
                            float* __restrict__ c_state,      // [128,512]
                            const unsigned short* __restrict__ whh,  // [2048,512] bf16
                            const float* __restrict__ gx,     // [16384,2048]
                            const int* __restrict__ done,     // [16384]
                            unsigned short* __restrict__ abuf, // [16384,544] bf16
                            int t) {
  int wave = blockIdx.x * (blockDim.x >> 5) + (threadIdx.x >> 5);
  int lane = threadIdx.x & 31;
  int tm = wave >> 5;
  int tn = wave & 31;
  int half = (lane >> 4) & 1;
  int mrow = tm << 4, ncol = tn << 4;
  int arow = mrow + (lane & 15);
  float am = 1.0f - (float)done[t * 128 + arow];
  v8f ai = {0.f,0.f,0.f,0.f,0.f,0.f,0.f,0.f};
  v8f af = ai, ag = ai, ao = ai;
  for (int kb = 0; kb < 512; kb += 32) {
    FragBF fa;
    const float* p = h_in + (size_t)arow * 512 + kb + half * 8;
#pragma unroll
    for (int e = 0; e < 8; ++e) fa.u[e] = f2bf(p[e] * am);
#pragma unroll
    for (int e = 0; e < 8; ++e) fa.u[8 + e] = f2bf(p[16 + e] * am);
    v16bf a = fa.v;
    ai = wmma_bf16(a, load_b_wt(whh,                    512, ncol, kb, lane), ai);
    af = wmma_bf16(a, load_b_wt(whh + (size_t)262144,   512, ncol, kb, lane), af);
    ag = wmma_bf16(a, load_b_wt(whh + (size_t)2*262144, 512, ncol, kb, lane), ag);
    ao = wmma_bf16(a, load_b_wt(whh + (size_t)3*262144, 512, ncol, kb, lane), ao);
  }
  int hid = ncol + (lane & 15);
#pragma unroll
  for (int v = 0; v < 8; ++v) {
    int b = mrow + v + (half << 3);
    int rtb = t * 128 + b;
    const float* g = gx + (size_t)rtb * 2048;
    float m = 1.0f - (float)done[rtb];
    float gi = ai[v] + g[hid];
    float gf = af[v] + g[512 + hid];
    float gg = ag[v] + g[1024 + hid];
    float go = ao[v] + g[1536 + hid];
    float si = 1.f / (1.f + expf(-gi));
    float sf = 1.f / (1.f + expf(-gf));
    float so = 1.f / (1.f + expf(-go));
    float tg = tanhf(gg);
    float cm = c_state[(size_t)b * 512 + hid] * m;
    float cn = sf * cm + si * tg;
    float hn = so * tanhf(cn);
    c_state[(size_t)b * 512 + hid] = cn;
    h_out[(size_t)b * 512 + hid] = hn;
    abuf[(size_t)rtb * 544 + hid] = f2bf(hn);
  }
}

// ---------------------------------------------------------------------------
// Heads: 7-way softmax policy + value, one thread per row (tiny FLOPs).
// out[r] = [log_prob(action), entropy, value, probs(7)]
// ---------------------------------------------------------------------------
__global__ void heads_k(const float* __restrict__ lf,  // [16384,128]
                        const float* __restrict__ aw,  // [7,128]
                        const float* __restrict__ ab,  // [7]
                        const float* __restrict__ cw,  // [128]
                        const float* __restrict__ cb,  // [1]
                        const int* __restrict__ action,
                        float* __restrict__ out) {
  int r = blockIdx.x * blockDim.x + threadIdx.x;
  if (r >= 16384) return;
  const float* f = lf + (size_t)r * 128;
  float lg[7];
#pragma unroll
  for (int j = 0; j < 7; ++j) lg[j] = ab[j];
  float val = cb[0];
  for (int k = 0; k < 128; ++k) {
    float x = f[k];
#pragma unroll
    for (int j = 0; j < 7; ++j) lg[j] = fmaf(aw[j * 128 + k], x, lg[j]);
    val = fmaf(cw[k], x, val);
  }
  float mx = lg[0];
#pragma unroll
  for (int j = 1; j < 7; ++j) mx = fmaxf(mx, lg[j]);
  float se = 0.f;
#pragma unroll
  for (int j = 0; j < 7; ++j) se += expf(lg[j] - mx);
  float lse = mx + logf(se);
  int a = action[r];
  float ent = 0.f, lpa = 0.f;
  float probs[7];
#pragma unroll
  for (int j = 0; j < 7; ++j) {
    float lp = lg[j] - lse;
    float p = expf(lp);
    probs[j] = p;
    ent -= p * lp;
    if (j == a) lpa = lp;
  }
  float* o = out + (size_t)r * 10;
  o[0] = lpa;
  o[1] = ent;
  o[2] = val;
#pragma unroll
  for (int j = 0; j < 7; ++j) o[3 + j] = probs[j];
}

// ---------------------------------------------------------------------------
// Host launcher
// ---------------------------------------------------------------------------
extern "C" void kernel_launch(void* const* d_in, const int* in_sizes, int n_in,
                              void* d_out, int out_size, void* d_ws, size_t ws_size,
                              hipStream_t stream) {
  (void)in_sizes; (void)n_in; (void)out_size; (void)ws_size;
  const float* x       = (const float*)d_in[0];
  const int*   done    = (const int*)  d_in[1];
  const int*   z       = (const int*)  d_in[2];
  const int*   action  = (const int*)  d_in[3];
  const float* h0      = (const float*)d_in[4];
  const float* c0      = (const float*)d_in[5];
  const float* conv1_w = (const float*)d_in[6];
  const float* conv1_b = (const float*)d_in[7];
  const float* conv2_w = (const float*)d_in[8];
  const float* conv2_b = (const float*)d_in[9];
  const float* conv3_w = (const float*)d_in[10];
  const float* conv3_b = (const float*)d_in[11];
  const float* fc1_w   = (const float*)d_in[12];
  const float* fc1_b   = (const float*)d_in[13];
  const float* fc2_w   = (const float*)d_in[14];
  const float* fc2_b   = (const float*)d_in[15];
  const float* w_ih    = (const float*)d_in[16];
  const float* w_hh    = (const float*)d_in[17];
  const float* b_ih    = (const float*)d_in[18];
  const float* b_hh    = (const float*)d_in[19];
  const float* lofeat_w= (const float*)d_in[20];
  const float* lofeat_b= (const float*)d_in[21];
  const float* actor_w = (const float*)d_in[22];
  const float* actor_b = (const float*)d_in[23];
  const float* critic_w= (const float*)d_in[24];
  const float* critic_b= (const float*)d_in[25];
  float* out = (float*)d_out;

  char* base = (char*)d_ws;
  size_t off = 0;
  auto bump = [&](size_t bytes) -> char* {
    char* p = base + off;
    off = (off + bytes + 255) & ~(size_t)255;
    return p;
  };
  unsigned short* B0  = (unsigned short*)bump((size_t)16384 * 4096 * 2);  // 134 MB
  unsigned short* B1  = (unsigned short*)bump((size_t)16384 * 4096 * 2);  // 134 MB
  unsigned short* H1  = (unsigned short*)bump((size_t)16384 * 512 * 2);
  unsigned short* H2  = (unsigned short*)bump((size_t)16384 * 512 * 2);
  unsigned short* WF1 = (unsigned short*)bump((size_t)512 * 4096 * 2);
  unsigned short* WF2 = (unsigned short*)bump((size_t)512 * 512 * 2);
  unsigned short* WIH = (unsigned short*)bump((size_t)2048 * 512 * 2);
  unsigned short* WHH = (unsigned short*)bump((size_t)2048 * 512 * 2);
  unsigned short* WLO = (unsigned short*)bump((size_t)128 * 544 * 2);
  unsigned short* WK1 = (unsigned short*)bump((size_t)(416 / 32) * 512 * 2);
  unsigned short* WK2 = (unsigned short*)bump((size_t)(160 / 32) * 512 * 2);
  unsigned short* WK3 = (unsigned short*)bump((size_t)(160 / 32) * 512 * 2);
  float* BSUM = (float*)bump(2048 * 4);
  float* HS0  = (float*)bump((size_t)128 * 512 * 4);
  float* HS1  = (float*)bump((size_t)128 * 512 * 4);
  float* CS   = (float*)bump((size_t)128 * 512 * 4);
  // Aliases: conv ping-pong buffers are dead once fc1 has consumed B1.
  float*          GX   = (float*)B0;                        // [16384,2048] f32
  unsigned short* ABUF = B1;                                // [16384,544] bf16
  float*          LF   = (float*)((char*)B1 + (size_t)16384 * 544 * 2);  // [16384,128]

  const int TPB = 256;
  // --- setup / conversions ---
  cvt_f32_bf16_k<<<(67108864 + TPB - 1) / TPB, TPB, 0, stream>>>(x, B0, 67108864L);
  cvt_f32_bf16_k<<<(2097152 + TPB - 1) / TPB, TPB, 0, stream>>>(fc1_w, WF1, 2097152L);
  cvt_f32_bf16_k<<<(262144 + TPB - 1) / TPB, TPB, 0, stream>>>(fc2_w, WF2, 262144L);
  cvt_f32_bf16_k<<<(1048576 + TPB - 1) / TPB, TPB, 0, stream>>>(w_ih, WIH, 1048576L);
  cvt_f32_bf16_k<<<(1048576 + TPB - 1) / TPB, TPB, 0, stream>>>(w_hh, WHH, 1048576L);
  pack_conv_w_k<<<((416 / 32) * 512 + TPB - 1) / TPB, TPB, 0, stream>>>(conv1_w, WK1, 400, 416);
  pack_conv_w_k<<<((160 / 32) * 512 + TPB - 1) / TPB, TPB, 0, stream>>>(conv2_w, WK2, 144, 160);
  pack_conv_w_k<<<((160 / 32) * 512 + TPB - 1) / TPB, TPB, 0, stream>>>(conv3_w, WK3, 144, 160);
  bias_sum_k<<<(2048 + TPB - 1) / TPB, TPB, 0, stream>>>(b_ih, b_hh, BSUM, 2048);
  prep_lofw_k<<<(128 * 544 + TPB - 1) / TPB, TPB, 0, stream>>>(lofeat_w, WLO);
  copy_f32_k<<<(65536 + TPB - 1) / TPB, TPB, 0, stream>>>(h0, HS0, 65536);
  copy_f32_k<<<(65536 + TPB - 1) / TPB, TPB, 0, stream>>>(c0, CS, 65536);

  // --- conv stack (implicit GEMM, WMMA): B0 -> B1 -> B0 -> B1 ---
  const int convBlocks = (16384 * 16) / 8;  // 1 wave per (img,row), 8 waves/block
  conv_wmma_k<25, 5, 2, 400, 416><<<convBlocks, TPB, 0, stream>>>(B0, WK1, conv1_b, B1);
  conv_wmma_k<9, 3, 1, 144, 160><<<convBlocks, TPB, 0, stream>>>(B1, WK2, conv2_b, B0);
  conv_wmma_k<9, 3, 1, 144, 160><<<convBlocks, TPB, 0, stream>>>(B0, WK3, conv3_b, B1);

  // --- fc1: [16384,4096] @ [512,4096]^T -> H1 (bf16, leaky) ---
  fc_gemm_k<<<(1024 * 8) / 8, TPB, 0, stream>>>(B1, 4096, WF1, 4096, fc1_b,
                                                nullptr, H1, 512,
                                                16384, 512, 4096, 1);
  // B1 is now free: set up one-hot / K-padding tail of ABUF (alias of B1)
  prep_abuf_tail_k<<<(16384 * 32 + TPB - 1) / TPB, TPB, 0, stream>>>(z, ABUF);

  // --- fc2: [16384,512] @ [512,512]^T -> H2 (bf16, leaky) ---
  fc_gemm_k<<<(1024 * 8) / 8, TPB, 0, stream>>>(H1, 512, WF2, 512, fc2_b,
                                                nullptr, H2, 512,
                                                16384, 512, 512, 1);
  // --- precompute input-side LSTM gates: GX = H2 @ W_ih^T + (b_ih+b_hh) ---
  fc_gemm_k<<<(1024 * 32) / 8, TPB, 0, stream>>>(H2, 512, WIH, 512, BSUM,
                                                 GX, nullptr, 0,
                                                 16384, 2048, 512, 0);
  // --- sequential LSTM scan: 128 fused GEMM+pointwise steps, h ping-pong ---
  for (int t = 0; t < 128; ++t) {
    const float* hin = (t & 1) ? HS1 : HS0;
    float* hout      = (t & 1) ? HS0 : HS1;
    lstm_step_k<<<32, TPB, 0, stream>>>(hin, hout, CS, WHH, GX, done, ABUF, t);
  }
  // --- lofeat: [16384,544] @ [128,544]^T -> LF (f32, leaky) ---
  fc_gemm_k<<<(1024 * 2) / 8, TPB, 0, stream>>>(ABUF, 544, WLO, 544, lofeat_b,
                                                LF, nullptr, 0,
                                                16384, 128, 544, 1);
  // --- heads ---
  heads_k<<<16384 / TPB, TPB, 0, stream>>>(LF, actor_w, actor_b, critic_w,
                                           critic_b, action, out);
}
